// CSMA_6820408066305
// MI455X (gfx1250) — matrix operand order repaired and checked by reading, not compiled, not run on hardware
//
#include <hip/hip_runtime.h>

// Problem constants (from reference)
#define Bn    8
#define Cn    128
#define Nn    16
#define HWn   3136
#define Ln    (Nn * HWn)       // 50176
#define CHUNK 64
#define NCHUNK (HWn / CHUNK)   // 49

typedef __attribute__((ext_vector_type(16))) __bf16 v16bf;
typedef __attribute__((ext_vector_type(8)))  float  v8f;

union BPack {
    unsigned int u[8];
    v16bf v;
};

__device__ __forceinline__ unsigned short f2bf(float f) {
    // round-to-nearest-even fp32 -> bf16
    unsigned int u = __float_as_uint(f);
    u += 0x7FFFu + ((u >> 16) & 1u);
    return (unsigned short)(u >> 16);
}

// ---------------------------------------------------------------------------
// Zero the atomically-accumulated region of the workspace (G + s).
// ---------------------------------------------------------------------------
__global__ void k_zero(float* __restrict__ p, int n) {
    int i = blockIdx.x * blockDim.x + threadIdx.x;
    if (i < n) p[i] = 0.0f;
}

// ---------------------------------------------------------------------------
// Single pass over x: per batch compute
//   G  = x x^T          (128x128, bf16 WMMA, fp32 accum, atomic reduce)
//   s  = x . 1          (row sums)
//   xbar = mean_N x     (128 x 3136)
// grid = Bn*NCHUNK blocks (one per (b, 64-wide hw chunk)), 256 threads.
// ---------------------------------------------------------------------------
__global__ __launch_bounds__(256) void k_stats(const float* __restrict__ x,
                                               float* __restrict__ G,
                                               float* __restrict__ s,
                                               float* __restrict__ xbar) {
    // bf16 tile, 128 rows x 64 cols, row stride 66 shorts (33 dwords) -> no bank conflicts
    __shared__ unsigned short shx[128 * 66];

    const int b     = blockIdx.x / NCHUNK;
    const int chunk = blockIdx.x % NCHUNK;
    const int hw0   = chunk * CHUNK;

    const int t    = threadIdx.x;
    const int col  = t & 63;
    const int crow = t >> 6;       // 0..3
    const int lane = t & 31;
    const int wv   = t >> 5;       // wave id 0..7 -> i-tile
    const int m    = lane & 15;
    const int half = lane >> 4;

    float xsum[32];
#pragma unroll
    for (int i = 0; i < 32; ++i) xsum[i] = 0.0f;

    v8f acc[8];
#pragma unroll
    for (int j = 0; j < 8; ++j) acc[j] = (v8f){0.f, 0.f, 0.f, 0.f, 0.f, 0.f, 0.f, 0.f};

    const float* xb = x + (size_t)b * Cn * Ln;

    for (int n = 0; n < Nn; ++n) {
        // ---- stage 128x64 fp32 -> bf16 tile; fold xbar/s accumulation in ----
#pragma unroll
        for (int i = 0; i < 32; ++i) {
            int c = crow + 4 * i;
            float v = xb[(size_t)c * Ln + (size_t)n * HWn + hw0 + col];
            xsum[i] += v;
            shx[c * 66 + col] = f2bf(v);
        }
        __syncthreads();

        // ---- WMMA: wave wv owns i-tile = wv, all 8 j-tiles, K = 64 cols ----
        const unsigned int* shu = (const unsigned int*)shx;
        const int arow = 16 * wv + m;
#pragma unroll
        for (int kc = 0; kc < 2; ++kc) {       // two K=32 chunks
            const int kb = kc * 16;            // dword offset within row
            BPack A;
#pragma unroll
            for (int v2 = 0; v2 < 4; ++v2) {
                A.u[v2]     = shu[arow * 33 + kb + half * 4 + v2];       // K = 2v(+8 if hi-half)
                A.u[4 + v2] = shu[arow * 33 + kb + 8 + half * 4 + v2];   // K = 16+...
            }
#pragma unroll
            for (int tj = 0; tj < 8; ++tj) {
                BPack Bm;
                const int brow = 16 * tj + m;  // B column index (channel j)
#pragma unroll
                for (int v2 = 0; v2 < 8; ++v2)
                    Bm.u[v2] = shu[brow * 33 + kb + (v2 >> 2) * 8 + half * 4 + (v2 & 3)];      // K = 2v(+8 if hi-half), +16 for v>=4
                acc[tj] = __builtin_amdgcn_wmma_f32_16x16x32_bf16(
                    false, A.v, false, Bm.v, (short)0, acc[tj], false, false);
            }
        }
        __syncthreads();
    }

    // ---- xbar + s ----
#pragma unroll
    for (int i = 0; i < 32; ++i) {
        int c = crow + 4 * i;
        xbar[((size_t)b * Cn + c) * HWn + hw0 + col] = xsum[i] * (1.0f / Nn);
        atomicAdd(&s[b * Cn + c], xsum[i]);
    }

    // ---- reduce partial G tiles into global ----
    float* Gb = G + (size_t)b * Cn * Cn;
#pragma unroll
    for (int tj = 0; tj < 8; ++tj) {
#pragma unroll
        for (int r = 0; r < 8; ++r) {
            int gi = 16 * wv + r + 8 * half;   // C/D layout: vgpr r -> M=r (lo) / M=8+r (hi)
            int gj = 16 * tj + m;
            atomicAdd(&Gb[gi * Cn + gj], acc[tj][r]);
        }
    }
}

// ---------------------------------------------------------------------------
// Tiny per-batch algebra chain (VALU; ~134 MFLOP total).
// ---------------------------------------------------------------------------
__global__ __launch_bounds__(1024) void k_qg(const float* __restrict__ G,
                                             const float* __restrict__ s,
                                             const float* __restrict__ wq,
                                             const float* __restrict__ wk,
                                             float* __restrict__ QG,
                                             float* __restrict__ u,
                                             float* __restrict__ v2) {
    const int b = blockIdx.x, t = threadIdx.x;
    const float* Gb = G + b * Cn * Cn;
    for (int e = t; e < Cn * Cn; e += 1024) {
        int o = e >> 7, k = e & 127;
        float a = 0.0f;
        for (int c = 0; c < Cn; ++c) a += wq[o * Cn + c] * Gb[c * Cn + k];
        QG[b * Cn * Cn + e] = a;
    }
    if (t < Cn) {
        float a = 0.0f, bb = 0.0f;
        for (int c = 0; c < Cn; ++c) {
            float sv = s[b * Cn + c];
            a  += wq[t * Cn + c] * sv;
            bb += wk[t * Cn + c] * sv;
        }
        u[b * Cn + t]  = a;
        v2[b * Cn + t] = bb;
    }
}

__global__ __launch_bounds__(1024) void k_softmax(const float* __restrict__ QG,
                                                  const float* __restrict__ wk,
                                                  const float* __restrict__ u,
                                                  const float* __restrict__ v2,
                                                  const float* __restrict__ bq,
                                                  const float* __restrict__ bk,
                                                  float* __restrict__ attn) {
    __shared__ float S[Cn * Cn];   // 64 KB
    const int b = blockIdx.x, t = threadIdx.x;
    const float* QGb = QG + b * Cn * Cn;
    for (int e = t; e < Cn * Cn; e += 1024) {
        int i = e >> 7, j = e & 127;
        float a = 0.0f;
        for (int k = 0; k < Cn; ++k) a += QGb[i * Cn + k] * wk[j * Cn + k];
        a += u[b * Cn + i] * bk[j] + bq[i] * v2[b * Cn + j] + (float)Ln * bq[i] * bk[j];
        S[e] = a;
    }
    __syncthreads();
    if (t < Cn) {
        float mx = -1e30f;
        for (int j = 0; j < Cn; ++j) mx = fmaxf(mx, S[t * Cn + j]);
        float sum = 0.0f;
        for (int j = 0; j < Cn; ++j) {
            float e2 = __expf(S[t * Cn + j] - mx);
            S[t * Cn + j] = e2;
            sum += e2;
        }
        float inv = 1.0f / sum;
        for (int j = 0; j < Cn; ++j) attn[b * Cn * Cn + t * Cn + j] = S[t * Cn + j] * inv;
    }
}

__global__ __launch_bounds__(1024) void k_av(const float* __restrict__ attn,
                                             const float* __restrict__ wv,
                                             const float* __restrict__ bv,
                                             float* __restrict__ AV,
                                             float* __restrict__ tvec) {
    const int b = blockIdx.x, t = threadIdx.x;
    const float* at = attn + b * Cn * Cn;
    for (int e = t; e < Cn * Cn; e += 1024) {
        int i = e >> 7, c = e & 127;
        float a = 0.0f;
        for (int k = 0; k < Cn; ++k) a += at[i * Cn + k] * wv[k * Cn + c];
        AV[b * Cn * Cn + e] = a;
    }
    if (t < Cn) {
        float a = 0.0f;
        for (int k = 0; k < Cn; ++k) a += at[t * Cn + k] * bv[k];
        tvec[b * Cn + t] = a;
    }
}

__global__ __launch_bounds__(1024) void k_m(const float* __restrict__ AV,
                                            const float* __restrict__ wo,
                                            const float* __restrict__ bo,
                                            const float* __restrict__ tvec,
                                            float* __restrict__ Mg,
                                            float* __restrict__ cvec) {
    const int b = blockIdx.x, t = threadIdx.x;
    const float* AVb = AV + b * Cn * Cn;
    for (int e = t; e < Cn * Cn; e += 1024) {
        int o = e >> 7, c = e & 127;
        float a = 0.0f;
        for (int i = 0; i < Cn; ++i) a += wo[o * Cn + i] * AVb[i * Cn + c];
        Mg[b * Cn * Cn + e] = a;
    }
    if (t < Cn) {
        float a = bo[t];
        for (int i = 0; i < Cn; ++i) a += wo[t * Cn + i] * tvec[b * Cn + i];
        cvec[b * Cn + t] = a;
    }
}

// ---------------------------------------------------------------------------
// out[b,o,hw] = (M_b xbar)[o,hw] + xbar[o,hw] + cvec[o]
// WMMA for M @ xbar (bf16 in / fp32 accum); residual kept in fp32.
// grid = Bn*NCHUNK, 256 threads.
// ---------------------------------------------------------------------------
__global__ __launch_bounds__(256) void k_out(const float* __restrict__ Mg,
                                             const float* __restrict__ cvec,
                                             const float* __restrict__ xbar,
                                             float* __restrict__ out) {
    __shared__ unsigned short shM[128 * 130];  // M bf16, row stride 130 shorts (65 dwords)
    __shared__ unsigned short shX[64 * 130];   // xbar tile transposed [col][c]

    const int b     = blockIdx.x / NCHUNK;
    const int chunk = blockIdx.x % NCHUNK;
    const int hw0   = chunk * CHUNK;

    const int t    = threadIdx.x;
    const int lane = t & 31;
    const int wv   = t >> 5;      // i-tile
    const int m    = lane & 15;
    const int half = lane >> 4;

    const float* Mb = Mg + b * Cn * Cn;
    const float* xb = xbar + (size_t)b * Cn * HWn;

    // stage M (bf16)
    for (int e = t; e < Cn * Cn; e += 256) {
        int o = e >> 7, c = e & 127;
        shM[o * 130 + c] = f2bf(Mb[e]);
    }
    // stage xbar chunk, transposed so K (=channel) pairs are contiguous
    {
        int col = t & 63, crow = t >> 6;
#pragma unroll
        for (int i = 0; i < 32; ++i) {
            int c = crow + 4 * i;
            shX[col * 130 + c] = f2bf(xb[(size_t)c * HWn + hw0 + col]);
        }
    }
    __syncthreads();

    v8f acc[4];
#pragma unroll
    for (int j = 0; j < 4; ++j) acc[j] = (v8f){0.f, 0.f, 0.f, 0.f, 0.f, 0.f, 0.f, 0.f};

    const unsigned int* mu = (const unsigned int*)shM;
    const unsigned int* xu = (const unsigned int*)shX;
    const int arow = 16 * wv + m;

#pragma unroll
    for (int kc = 0; kc < 4; ++kc) {           // K = 128 in chunks of 32
        const int kb = kc * 16;
        BPack A;
#pragma unroll
        for (int v2 = 0; v2 < 4; ++v2) {
            A.u[v2]     = mu[arow * 65 + kb + half * 4 + v2];
            A.u[4 + v2] = mu[arow * 65 + kb + 8 + half * 4 + v2];
        }
#pragma unroll
        for (int tj = 0; tj < 4; ++tj) {
            BPack Bm;
            const int bcol = 16 * tj + m;
#pragma unroll
            for (int v2 = 0; v2 < 8; ++v2)
                Bm.u[v2] = xu[bcol * 65 + kb + (v2 >> 2) * 8 + half * 4 + (v2 & 3)];
            acc[tj] = __builtin_amdgcn_wmma_f32_16x16x32_bf16(
                false, A.v, false, Bm.v, (short)0, acc[tj], false, false);
        }
    }

    // store with fp32 residual + bias
    float* ob = out + (size_t)b * Cn * HWn;
#pragma unroll
    for (int tj = 0; tj < 4; ++tj) {
#pragma unroll
        for (int r = 0; r < 8; ++r) {
            int o = 16 * wv + r + 8 * half;
            int colhw = hw0 + 16 * tj + m;
            size_t idx = (size_t)o * HWn + colhw;
            ob[idx] = acc[tj][r] + xb[idx] + cvec[b * Cn + o];
        }
    }
}

// ---------------------------------------------------------------------------
extern "C" void kernel_launch(void* const* d_in, const int* in_sizes, int n_in,
                              void* d_out, int out_size, void* d_ws, size_t ws_size,
                              hipStream_t stream) {
    const float* x  = (const float*)d_in[0];
    const float* wq = (const float*)d_in[1];
    const float* bq = (const float*)d_in[2];
    const float* wk = (const float*)d_in[3];
    const float* bk = (const float*)d_in[4];
    const float* wv = (const float*)d_in[5];
    const float* bv = (const float*)d_in[6];
    const float* wo = (const float*)d_in[7];
    const float* bo = (const float*)d_in[8];
    float* out = (float*)d_out;

    float* ws   = (float*)d_ws;
    float* G    = ws;                 // 8*128*128 = 131072
    float* s    = G    + 131072;      // 1024
    float* u    = s    + 1024;        // 1024
    float* v2   = u    + 1024;        // 1024
    float* tvec = v2   + 1024;        // 1024
    float* cvec = tvec + 1024;        // 1024
    float* QG   = cvec + 1024;        // 131072
    float* attn = QG   + 131072;      // 131072
    float* AV   = attn + 131072;      // 131072
    float* Mm   = AV   + 131072;      // 131072
    float* xbar = Mm   + 131072;      // 8*128*3136 = 3211264  (total ~15.5 MB)

    (void)in_sizes; (void)n_in; (void)out_size; (void)ws_size;

    k_zero<<<(132096 + 255) / 256, 256, 0, stream>>>(ws, 132096);  // G + s
    k_stats<<<Bn * NCHUNK, 256, 0, stream>>>(x, G, s, xbar);
    k_qg<<<Bn, 1024, 0, stream>>>(G, s, wq, wk, QG, u, v2);
    k_softmax<<<Bn, 1024, 0, stream>>>(QG, wk, u, v2, bq, bk, attn);
    k_av<<<Bn, 1024, 0, stream>>>(attn, wv, bv, AV, tvec);
    k_m<<<Bn, 1024, 0, stream>>>(AV, wo, bo, tvec, Mm, cvec);
    k_out<<<Bn * NCHUNK, 256, 0, stream>>>(Mm, cvec, xbar, out);
}
